// SubLSTMCudaCell_8478265442721
// MI455X (gfx1250) — matrix operand order repaired and checked by reading, not compiled, not run on hardware
//
#include <hip/hip_runtime.h>
#include <stdint.h>

// ---------------------------------------------------------------------------
// subLSTM cell, fused bf16-WMMA implementation for gfx1250 (MI455X).
//   gates = sigmoid([old_h|input] @ W^T + bias)   (8192 x 4096, K=2048)
//   new_cell = old_cell*f + z - i ; new_h = sigmoid(new_cell) - o
// GEMM in bf16 / f32-accum via v_wmma_f32_16x16x32_bf16.
// LDS tiles double-buffered, filled with global_load_async_to_lds_b128
// (ASYNCcnt pipeline), drained with ds_load_b128.
// Block tile: 128 batch rows x 32 h-cols x 4 gates (= 128x128 gate tile).
// 8 waves as 4(M) x 2(N); each wave: 2 M-frags x 1 N-frag x 4 gates
// -> 8 accumulators, 8 WMMAs per 16x16x32 k-step (A reused x4, B reused x2).
// ---------------------------------------------------------------------------

typedef __attribute__((ext_vector_type(16))) __bf16 v16bf;
typedef __attribute__((ext_vector_type(8)))  __bf16 v8bf;
typedef __attribute__((ext_vector_type(8)))  float  v8f;

#define B_DIM 8192
#define H_DIM 1024
#define I_DIM 1024
#define K_DIM 2048          // H + I
#define G_DIM 4096          // 4H

#define BM 128              // batch rows per block
#define BNH 32              // h-columns per block (x4 gates -> 128 gate cols)
#define BK 64               // K elements (bf16) per pipeline stage
#define NSTAGE (K_DIM / BK) // 32

static __device__ __forceinline__ uint16_t f2bf(float f) {
    union { float f; uint32_t u; } v;
    v.f = f;
    uint32_t r = v.u + 0x7FFFu + ((v.u >> 16) & 1u);   // round-to-nearest-even
    return (uint16_t)(r >> 16);
}

static __device__ __forceinline__ float sigmoidf(float x) {
    return 1.0f / (1.0f + __expf(-x));
}

// A-fragment: lane half 0 holds K {0..7,16..23}, half 1 holds {8..15,24..31};
// p already includes the half*8 offset, second chunk is +16 halfs.
static __device__ __forceinline__ v16bf load_afrag(const uint16_t* p) {
    const v8bf lo = *(const v8bf*)(p);
    const v8bf hi = *(const v8bf*)(p + 16);
    v16bf a;
    #pragma unroll
    for (int e = 0; e < 8; ++e) { a[e] = lo[e]; a[e + 8] = hi[e]; }
    return a;
}

// --------------------------- prep: fp32 -> bf16 ----------------------------
// W16[g][k] = bf16(weights[g][k])                              (4096 x 2048)
// X16[b][k] = bf16(k < 1024 ? old_h[b][k] : input[b][k-1024])  (8192 x 2048)
__global__ void prep_pack(const float* __restrict__ input,
                          const float* __restrict__ weights,
                          const float* __restrict__ old_h,
                          uint16_t* __restrict__ W16,
                          uint16_t* __restrict__ X16) {
    const size_t NW4 = (size_t)G_DIM * K_DIM / 4;   // 4-element chunks
    const size_t NX4 = (size_t)B_DIM * K_DIM / 4;
    const size_t stride = (size_t)gridDim.x * blockDim.x;
    for (size_t i = (size_t)blockIdx.x * blockDim.x + threadIdx.x;
         i < NW4 + NX4; i += stride) {
        float4 v;
        uint16_t* dst;
        if (i < NW4) {
            v   = ((const float4*)weights)[i];
            dst = W16 + i * 4;
        } else {
            const size_t j   = i - NW4;            // chunk index into X16
            const size_t row = (j * 4) / K_DIM;
            const size_t k   = (j * 4) % K_DIM;    // chunk never crosses k=1024
            const float* src = (k < (size_t)H_DIM)
                                   ? (old_h + row * H_DIM + k)
                                   : (input + row * I_DIM + (k - H_DIM));
            v   = *(const float4*)src;
            dst = X16 + j * 4;
        }
        const uint64_t packed = (uint64_t)f2bf(v.x)
                              | ((uint64_t)f2bf(v.y) << 16)
                              | ((uint64_t)f2bf(v.z) << 32)
                              | ((uint64_t)f2bf(v.w) << 48);
        *(uint64_t*)dst = packed;
    }
}

// ------------------------------ fused GEMM ---------------------------------
__global__ void __launch_bounds__(256)
sublstm_gemm(const uint16_t* __restrict__ X16,
             const uint16_t* __restrict__ W16,
             const float* __restrict__ bias,
             const float* __restrict__ old_cell,
             float* __restrict__ out) {   // [0,8M): new_h  [8M,16M): new_cell
    __shared__ __align__(128) uint16_t sA[2][BM * BK];        // 16 KB each
    __shared__ __align__(128) uint16_t sB[2][4 * BNH * BK];   // 16 KB each

    const int tid   = threadIdx.x;
    const int lane  = tid & 31;
    const int wave  = tid >> 5;
    const int wm    = wave >> 1;           // 0..3 : 32-row M strip
    const int wn    = wave & 1;            // 0..1 : 16-col N strip
    const int hTile = blockIdx.x;          // 0..31
    const int bTile = blockIdx.y;          // 0..63
    const int h0      = hTile * BNH;
    const int rowBase = bTile * BM;

    const uint16_t* gX = X16 + (size_t)rowBase * K_DIM;

    // Issue one pipeline stage of async global->LDS copies (8 instrs/wave).
    auto issue = [&](int s, int buf) {
        const int kb = s * BK;
        // A tile: 128 rows x 64 bf16 = 1024 x 16B chunks
        #pragma unroll
        for (int i = 0; i < 4; ++i) {
            const int c = tid + i * 256;
            const int r = c >> 3, c8 = c & 7;
            const uint32_t lds = (uint32_t)(uintptr_t)&sA[buf][r * BK + c8 * 8];
            const uint64_t ga  = (uint64_t)(uintptr_t)
                (gX + (size_t)r * K_DIM + kb + c8 * 8);
            asm volatile("global_load_async_to_lds_b128 %0, %1, off"
                         :: "v"(lds), "v"(ga) : "memory");
        }
        // B tile: 128 gate-rows (4 gates x 32 h-cols) x 64 bf16 = 1024 chunks
        #pragma unroll
        for (int i = 0; i < 4; ++i) {
            const int c = tid + i * 256;
            const int r = c >> 3, c8 = c & 7;
            const int gate = r >> 5, hc = r & 31;
            const uint32_t lds = (uint32_t)(uintptr_t)&sB[buf][r * BK + c8 * 8];
            const uint64_t ga  = (uint64_t)(uintptr_t)
                (W16 + (size_t)(gate * H_DIM + h0 + hc) * K_DIM + kb + c8 * 8);
            asm volatile("global_load_async_to_lds_b128 %0, %1, off"
                         :: "v"(lds), "v"(ga) : "memory");
        }
    };

    v8f acc[4][2] = {};                     // [gate][m-frag] f32 accumulators
    const int m    = lane & 15;             // row/col within fragment
    const int half = lane >> 4;             // K-half select per WMMA layout

    issue(0, 0);
    for (int s = 0; s < NSTAGE; ++s) {
        const int buf = s & 1;
        if (s + 1 < NSTAGE) {
            issue(s + 1, buf ^ 1);
            asm volatile("s_wait_asynccnt 8" ::: "memory"); // stage s complete
        } else {
            asm volatile("s_wait_asynccnt 0" ::: "memory");
        }
        __syncthreads();

        const uint16_t* aRow = &sA[buf][(wm * 32 + m) * BK];
        #pragma unroll
        for (int t = 0; t < 2; ++t) {       // two 16x16x32 K-steps per stage
            const int kk = t * 32;
            // Batch all fragment loads, then burst 8 WMMAs back-to-back.
            const v16bf a0 = *(const v16bf*)(aRow + kk + half * 16);
            const v16bf a1 = *(const v16bf*)(aRow + 16 * BK + kk + half * 16);
            v16bf bf[4];
            #pragma unroll
            for (int g = 0; g < 4; ++g)     // B: half 0 -> K 0..15, half 1 -> 16..31
                bf[g] = *(const v16bf*)
                    (&sB[buf][(g * BNH + wn * 16 + m) * BK + kk + half * 16]);
            #pragma unroll
            for (int g = 0; g < 4; ++g) {
                acc[g][0] = __builtin_amdgcn_wmma_f32_16x16x32_bf16(
                    false, a0, false, bf[g], (short)0, acc[g][0], false, false);
                acc[g][1] = __builtin_amdgcn_wmma_f32_16x16x32_bf16(
                    false, a1, false, bf[g], (short)0, acc[g][1], false, false);
            }
        }
        __syncthreads();                    // protect buffer refilled next iter
    }

    // ------------------------------ epilogue -------------------------------
    const int col = h0 + wn * 16 + m;
    float bb[4];
    #pragma unroll
    for (int g = 0; g < 4; ++g) bb[g] = bias[g * H_DIM + col];

    #pragma unroll
    for (int mf = 0; mf < 2; ++mf) {
        #pragma unroll
        for (int j = 0; j < 8; ++j) {       // C layout: M = j + 8*half, N = m
            const int b = rowBase + wm * 32 + mf * 16 + half * 8 + j;
            const float ing  = sigmoidf(acc[0][mf][j] + bb[0]);
            const float outg = sigmoidf(acc[1][mf][j] + bb[1]);
            const float zg   = sigmoidf(acc[2][mf][j] + bb[2]);
            const float fg   = sigmoidf(acc[3][mf][j] + bb[3]);
            const float cold  = old_cell[(size_t)b * H_DIM + col];
            const float ncell = cold * fg + zg - ing;
            const float nh    = sigmoidf(ncell) - outg;
            out[(size_t)b * H_DIM + col] = nh;
            out[(size_t)B_DIM * H_DIM + (size_t)b * H_DIM + col] = ncell;
        }
    }
}

// ------------------------------- launcher ----------------------------------
extern "C" void kernel_launch(void* const* d_in, const int* in_sizes, int n_in,
                              void* d_out, int out_size, void* d_ws, size_t ws_size,
                              hipStream_t stream) {
    const float* input    = (const float*)d_in[0];
    const float* weights  = (const float*)d_in[1];
    const float* bias     = (const float*)d_in[2];
    const float* old_h    = (const float*)d_in[3];
    const float* old_cell = (const float*)d_in[4];

    uint16_t* W16 = (uint16_t*)d_ws;                       // 16 MB
    uint16_t* X16 = W16 + (size_t)G_DIM * K_DIM;           // 32 MB (48 MB total)

    prep_pack<<<dim3(2048), dim3(256), 0, stream>>>(input, weights, old_h, W16, X16);

    sublstm_gemm<<<dim3(H_DIM / BNH, B_DIM / BM), dim3(256), 0, stream>>>(
        X16, W16, bias, old_cell, (float*)d_out);
}
